// PathModel3D_18476949307410
// MI455X (gfx1250) — matrix operand order repaired and checked by reading, not compiled, not run on hardware
//
#include <hip/hip_runtime.h>
#include <math.h>

// ---------------------------------------------------------------------------
// MI455X (gfx1250, wave32) implementation of the 5-layer path-transformer.
// All big matmuls + attention use v_wmma_f32_16x16x32_f16 (fp32 accumulate).
// Degenerate shapes (K=3 encoder entry, N=3 head) use dedicated VALU kernels
// so the WMMA GEMM is completely guard-free.
// ---------------------------------------------------------------------------

typedef __attribute__((ext_vector_type(16))) _Float16 v16h;
typedef __attribute__((ext_vector_type(8)))  float    v8f;
typedef __attribute__((ext_vector_type(4)))  _Float16 h4;

// A-matrix fragment K index (16-bit A 16x32, ISA 7.12.2):
//   contiguous 8-element chunks per lane-half -> compiles to 2x ds_load_b128.
__device__ __forceinline__ int frag_ka(int e, int half) {
    int v = e >> 1, p = e & 1;
    return ((v & 4) << 2) + (half << 3) + ((v & 3) << 1) + p;
}
// B-matrix fragment K index (same 8-element chunk order as frag_ka)
__device__ __forceinline__ int frag_kb(int e, int half) {
    return ((e & 8) << 1) + (half << 3) + (e & 7);
}

__device__ __forceinline__ h4 cvt_h4(float4 v) {
    h4 r = {(_Float16)v.x, (_Float16)v.y, (_Float16)v.z, (_Float16)v.w};
    return r;
}

// ---------------------------------------------------------------------------
// Guard-free GEMM: C[M,N] = act(A[M,K] @ W[N,K]^T + bias[N]) (+ R[M,N])
// Preconditions: M%64==0, N%64==0, K%32==0, lda/ldw%4==0 (enforced by host).
// block = 128 threads (4 waves); block tile 64x64; K step 32.
// Staging issues all 8 global b128 loads before converting (loads overlap).
// ---------------------------------------------------------------------------
#define SAP 36  // padded LDS row stride (f16 elems); 72B rows keep h4 8B-aligned

__global__ void __launch_bounds__(128)
k_gemm(const float* __restrict__ A, int lda,
       const float* __restrict__ W, int ldw,
       const float* __restrict__ bias,
       const float* __restrict__ R, int ldr,
       float* __restrict__ C, int ldc,
       int K, int relu)
{
    extern __shared__ char smem[];
    _Float16* sA = (_Float16*)smem;                       // [64][SAP]
    _Float16* sW = (_Float16*)(smem + 64 * SAP * 2);      // [64][SAP]

    const int tid  = threadIdx.x;
    const int wid  = tid >> 5;
    const int lane = tid & 31;
    const int half = lane >> 4;
    const int lrow = lane & 15;
    const int m0 = blockIdx.y * 64;
    const int n0 = blockIdx.x * 64;

    v8f acc[4];
    #pragma unroll
    for (int j = 0; j < 4; ++j)
        #pragma unroll
        for (int e = 0; e < 8; ++e) acc[j][e] = 0.0f;

    for (int k0 = 0; k0 < K; k0 += 32) {
        // ---- stage: batch all 8 b128 loads first, then convert + store ----
        float4 va[4], vw[4];
        #pragma unroll
        for (int it = 0; it < 4; ++it) {
            int s = tid + it * 128;
            int r = s >> 3, c4 = (s & 7) << 2;
            va[it] = *(const float4*)(A + (size_t)(m0 + r) * lda + k0 + c4);
            vw[it] = *(const float4*)(W + (size_t)(n0 + r) * ldw + k0 + c4);
        }
        #pragma unroll
        for (int it = 0; it < 4; ++it) {
            int s = tid + it * 128;
            int r = s >> 3, c4 = (s & 7) << 2;
            *(h4*)&sA[r * SAP + c4] = cvt_h4(va[it]);
            *(h4*)&sW[r * SAP + c4] = cvt_h4(vw[it]);
        }
        __syncthreads();

        // speculative prefetch of next K tile (global_prefetch_b8)
        if (k0 + 32 < K) {
            int r = tid >> 1, pc = (tid & 1) << 4;
            __builtin_prefetch(&A[(size_t)(m0 + r) * lda + k0 + 32 + pc]);
            __builtin_prefetch(&W[(size_t)(n0 + r) * ldw + k0 + 32 + pc]);
        }

        v16h af, bf;
        const int am = (wid << 4) + lrow;
        #pragma unroll
        for (int e = 0; e < 16; ++e) af[e] = sA[am * SAP + frag_ka(e, half)];
        #pragma unroll
        for (int nt = 0; nt < 4; ++nt) {
            const int bn = (nt << 4) + lrow;
            #pragma unroll
            for (int e = 0; e < 16; ++e) bf[e] = sW[bn * SAP + frag_kb(e, half)];
            acc[nt] = __builtin_amdgcn_wmma_f32_16x16x32_f16(
                false, af, false, bf, (short)0, acc[nt], false, false);
        }
        __syncthreads();
    }

    // epilogue: C layout — lane holds column n=lane&15, rows m = r + 8*(lane>>4)
    #pragma unroll
    for (int nt = 0; nt < 4; ++nt) {
        const int gn = n0 + (nt << 4) + lrow;
        const float bv = bias ? bias[gn] : 0.0f;
        #pragma unroll
        for (int r = 0; r < 8; ++r) {
            const int gm = m0 + (wid << 4) + r + (half << 3);
            float v = acc[nt][r] + bv;
            if (R) v += R[(size_t)gm * ldr + gn];
            if (relu) v = v > 0.0f ? v : 0.0f;
            C[(size_t)gm * ldc + gn] = v;
        }
    }
}

// ---------------------------------------------------------------------------
// Encoder layer 1: out[p, o] = relu(sum_{c<3} shape[p,c] * w[o,c] + b[o])
// K=3 makes WMMA pointless (29/32 of the slab would be zeros) -> VALU kernel.
// ---------------------------------------------------------------------------
__global__ void __launch_bounds__(256)
k_enc1(const float* __restrict__ shape, const float* __restrict__ w,
       const float* __restrict__ b, float* __restrict__ out, int M)
{
    int i = blockIdx.x * 256 + threadIdx.x;   // over M*64
    if (i >= M * 64) return;
    int row = i >> 6, o = i & 63;
    float p0 = shape[row * 3 + 0], p1 = shape[row * 3 + 1], p2 = shape[row * 3 + 2];
    float v = fmaf(w[o * 3 + 0], p0,
              fmaf(w[o * 3 + 1], p1,
              fmaf(w[o * 3 + 2], p2, b[o])));
    out[(size_t)row * 64 + o] = fmaxf(v, 0.0f);
}

// ---------------------------------------------------------------------------
// Output head (N=3): wave-per-row GEMV, shuffle reduction.
// ---------------------------------------------------------------------------
__global__ void __launch_bounds__(128)
k_head(const float* __restrict__ X, const float* __restrict__ W,
       const float* __restrict__ bias, float* __restrict__ out, int D)
{
    int row  = blockIdx.x * 4 + (threadIdx.x >> 5);
    int lane = threadIdx.x & 31;
    const float* xr = X + (size_t)row * D;
    float a0 = 0.0f, a1 = 0.0f, a2 = 0.0f;
    for (int c = lane; c < D; c += 32) {
        float xv = xr[c];
        a0 = fmaf(xv, W[c], a0);
        a1 = fmaf(xv, W[D + c], a1);
        a2 = fmaf(xv, W[2 * D + c], a2);
    }
    for (int o = 16; o; o >>= 1) {
        a0 += __shfl_xor(a0, o);
        a1 += __shfl_xor(a1, o);
        a2 += __shfl_xor(a2, o);
    }
    if (lane == 0) {
        out[(size_t)row * 3 + 0] = a0 + bias[0];
        out[(size_t)row * 3 + 1] = a1 + bias[1];
        out[(size_t)row * 3 + 2] = a2 + bias[2];
    }
}

// ---------------------------------------------------------------------------
// Fused attention for one (16-query tile, head, batch).
// Scores strip (16 x Lk) kept in LDS; softmax in LDS; PV via WMMA.
// ---------------------------------------------------------------------------
#define QP 72  // padded f16 row stride (144B rows keep h4 8B-aligned)

__global__ void __launch_bounds__(128)
k_attn(const float* __restrict__ Q, int ldq,
       const float* __restrict__ KV, int ldkv,
       int koff, int voff,
       const unsigned char* __restrict__ mask,
       int causal,
       float* __restrict__ O, int ldo,
       int Tq, int Lk)
{
    extern __shared__ char smem[];
    float*    sS   = (float*)smem;                             // [16][Lk] f32
    _Float16* sP   = (_Float16*)(smem + (size_t)16 * Lk * 4);  // [16][Lk] f16
    _Float16* sQ   = sP + (size_t)16 * Lk;                     // [16][QP]
    _Float16* sK   = sQ + 16 * QP;                             // 4 x [16][QP]
    _Float16* sV   = sK + 4 * 16 * QP;                         // [32][QP]
    float*    sRed = (float*)(sV + 32 * QP);                   // [16][8]
    float*    sMax = sRed + 16 * 8;                            // [16]
    float*    sSum = sMax + 16;                                // [16]

    const int tid  = threadIdx.x;
    const int wid  = tid >> 5;
    const int lane = tid & 31;
    const int half = lane >> 4;
    const int lrow = lane & 15;
    const int b = blockIdx.z, h = blockIdx.y;
    const int q0 = blockIdx.x * 16;
    const size_t qbase = (size_t)b * Tq;
    const float scale = 0.125f;  // 1/sqrt(64)

    // ---- load Q tile (16 x 64) -> f16 LDS, batched b128 loads
    {
        float4 v[2];
        #pragma unroll
        for (int it = 0; it < 2; ++it) {
            int s = tid + it * 128;           // 256 float4 slots
            int r = s >> 4, c4 = (s & 15) << 2;
            v[it] = *(const float4*)(Q + (qbase + q0 + r) * (size_t)ldq + h * 64 + c4);
        }
        #pragma unroll
        for (int it = 0; it < 2; ++it) {
            int s = tid + it * 128;
            int r = s >> 4, c4 = (s & 15) << 2;
            *(h4*)&sQ[r * QP + c4] = cvt_h4(v[it]);
        }
    }
    __syncthreads();

    v16h aq0, aq1;
    #pragma unroll
    for (int e = 0; e < 16; ++e) {
        int ka = frag_ka(e, half);
        aq0[e] = sQ[lrow * QP + ka];
        aq1[e] = sQ[lrow * QP + 32 + ka];
    }

    // ---- phase 1: S = scale * Q K^T (+mask), per-wave key blocks of 16
    const int nkb = Lk >> 4;
    _Float16* myK = sK + wid * 16 * QP;
    for (int j = wid; j < nkb; j += 4) {
        float4 v[8];
        #pragma unroll
        for (int it = 0; it < 8; ++it) {
            int s = lane + it * 32;            // 256 float4 slots per 16x64 tile
            int r = s >> 4, c4 = (s & 15) << 2;
            v[it] = *(const float4*)(
                KV + ((size_t)b * Lk + j * 16 + r) * (size_t)ldkv + koff + h * 64 + c4);
        }
        #pragma unroll
        for (int it = 0; it < 8; ++it) {
            int s = lane + it * 32;
            int r = s >> 4, c4 = (s & 15) << 2;
            *(h4*)&myK[r * QP + c4] = cvt_h4(v[it]);
        }
        v16h bf0, bf1;
        #pragma unroll
        for (int e = 0; e < 16; ++e) {
            int kb = frag_kb(e, half);
            bf0[e] = myK[lrow * QP + kb];
            bf1[e] = myK[lrow * QP + 32 + kb];
        }
        v8f s;
        #pragma unroll
        for (int e = 0; e < 8; ++e) s[e] = 0.0f;
        s = __builtin_amdgcn_wmma_f32_16x16x32_f16(false, aq0, false, bf0, (short)0, s, false, false);
        s = __builtin_amdgcn_wmma_f32_16x16x32_f16(false, aq1, false, bf1, (short)0, s, false, false);

        int kcol = j * 16 + lrow;
        float mterm = 0.0f;
        if (mask && mask[(size_t)b * Lk + kcol]) mterm = -1e9f;
        #pragma unroll
        for (int r = 0; r < 8; ++r) {
            int qm = r + (half << 3);
            float vv = s[r] * scale + mterm;
            if (causal && kcol > q0 + qm) vv = -1e9f;
            sS[qm * Lk + kcol] = vv;
        }
    }
    __syncthreads();

    // ---- phase 2: row softmax over Lk (8 threads per row)
    {
        int row = tid >> 3, sub = tid & 7;
        float mx = -3.0e38f;
        for (int c = sub; c < Lk; c += 8) mx = fmaxf(mx, sS[row * Lk + c]);
        sRed[row * 8 + sub] = mx;
        __syncthreads();
        if (sub == 0) {
            float m2 = sRed[row * 8];
            for (int i = 1; i < 8; ++i) m2 = fmaxf(m2, sRed[row * 8 + i]);
            sMax[row] = m2;
        }
        __syncthreads();
        float rm = sMax[row], sum = 0.0f;
        for (int c = sub; c < Lk; c += 8) {
            float p = __expf(sS[row * Lk + c] - rm);
            sP[row * Lk + c] = (_Float16)p;
            sum += p;
        }
        sRed[row * 8 + sub] = sum;
        __syncthreads();
        if (sub == 0) {
            float s2 = 0.0f;
            for (int i = 0; i < 8; ++i) s2 += sRed[row * 8 + i];
            sSum[row] = s2;
        }
        __syncthreads();
    }

    // ---- phase 3: O = P V ; wave `wid` owns dim slice [wid*16, wid*16+16)
    v8f oacc;
    #pragma unroll
    for (int e = 0; e < 8; ++e) oacc[e] = 0.0f;
    const int nvb = Lk >> 5;
    for (int kb = 0; kb < nvb; ++kb) {
        float4 v[4];
        #pragma unroll
        for (int it = 0; it < 4; ++it) {
            int s = tid + it * 128;            // 512 float4 slots per 32x64 tile
            int r = s >> 4, c4 = (s & 15) << 2;
            v[it] = *(const float4*)(
                KV + ((size_t)b * Lk + kb * 32 + r) * (size_t)ldkv + voff + h * 64 + c4);
        }
        #pragma unroll
        for (int it = 0; it < 4; ++it) {
            int s = tid + it * 128;
            int r = s >> 4, c4 = (s & 15) << 2;
            *(h4*)&sV[r * QP + c4] = cvt_h4(v[it]);
        }
        __syncthreads();
        v16h ap, bv;
        #pragma unroll
        for (int e = 0; e < 16; ++e) {
            int ka = frag_ka(e, half);
            int kb2 = frag_kb(e, half);
            ap[e] = sP[lrow * Lk + kb * 32 + ka];
            bv[e] = sV[kb2 * QP + (wid << 4) + lrow];
        }
        oacc = __builtin_amdgcn_wmma_f32_16x16x32_f16(false, ap, false, bv, (short)0, oacc, false, false);
        __syncthreads();
    }
    const int dn = (wid << 4) + lrow;
    #pragma unroll
    for (int r = 0; r < 8; ++r) {
        int qm = r + (half << 3);
        O[(qbase + q0 + qm) * (size_t)ldo + h * 64 + dn] = oacc[r] / sSum[qm];
    }
}

// ---------------------------------------------------------------------------
// Positional encoding  x[b,t,d]
// ---------------------------------------------------------------------------
__global__ void k_posenc(float* __restrict__ x, int T, int D, int B)
{
    int idx = blockIdx.x * blockDim.x + threadIdx.x;
    if (idx >= T * D) return;
    int t = idx / D, d = idx - t * D;
    int j = d >> 1;
    float freq = __expf(-(float)(2 * j) * (logf(10000.0f) / (float)D));
    float a = (float)t * freq;
    float v = (d & 1) ? __cosf(a) : __sinf(a);
    for (int b = 0; b < B; ++b)
        x[((size_t)b * T + t) * D + d] = v;
}

// ---------------------------------------------------------------------------
// Row LayerNorm, D = multiple of 512, block 128, float4 vectorized
// ---------------------------------------------------------------------------
__global__ void __launch_bounds__(128)
k_ln(const float* __restrict__ X, const float* __restrict__ g,
     const float* __restrict__ bta, float* __restrict__ Y, int D)
{
    __shared__ float rs[4], rss[4];
    const int row = blockIdx.x, tid = threadIdx.x, wid = tid >> 5;
    const float* xr = X + (size_t)row * D;
    float s = 0.0f, ss = 0.0f;
    for (int c4 = tid * 4; c4 < D; c4 += 512) {
        float4 v = *(const float4*)(xr + c4);
        s  += v.x + v.y + v.z + v.w;
        ss += v.x * v.x + v.y * v.y + v.z * v.z + v.w * v.w;
    }
    for (int o = 16; o; o >>= 1) {
        s  += __shfl_xor(s, o);
        ss += __shfl_xor(ss, o);
    }
    if ((tid & 31) == 0) { rs[wid] = s; rss[wid] = ss; }
    __syncthreads();
    float S = rs[0] + rs[1] + rs[2] + rs[3];
    float SS = rss[0] + rss[1] + rss[2] + rss[3];
    float mu = S / D, var = SS / D - mu * mu;
    float inv = rsqrtf(var + 1e-5f);
    float* yr = Y + (size_t)row * D;
    for (int c4 = tid * 4; c4 < D; c4 += 512) {
        float4 v = *(const float4*)(xr + c4);
        float4 gg = *(const float4*)(g + c4);
        float4 bb = *(const float4*)(bta + c4);
        float4 o;
        o.x = (v.x - mu) * inv * gg.x + bb.x;
        o.y = (v.y - mu) * inv * gg.y + bb.y;
        o.z = (v.z - mu) * inv * gg.z + bb.z;
        o.w = (v.w - mu) * inv * gg.w + bb.w;
        *(float4*)(yr + c4) = o;
    }
}

// ---------------------------------------------------------------------------
// path = start + cumsum(vel, axis=t)
// ---------------------------------------------------------------------------
__global__ void k_path(const float* __restrict__ vel, const float* __restrict__ start,
                       float* __restrict__ path, int T)
{
    int b = blockIdx.x, c = threadIdx.x;
    if (c >= 3) return;
    float acc = start[b * 3 + c];
    for (int t = 0; t < T; ++t) {
        acc += vel[((size_t)b * T + t) * 3 + c];
        path[((size_t)b * T + t) * 3 + c] = acc;
    }
}

// ---------------------------------------------------------------------------
// occupancy[b,n] = clip(sum_t exp(-100*|shape-path|^2), 0, 1)
// ---------------------------------------------------------------------------
__global__ void __launch_bounds__(256)
k_occ(const float* __restrict__ shape, const float* __restrict__ path,
      float* __restrict__ occ, int T, int N)
{
    __shared__ float sp[512 * 3];
    int b = blockIdx.y;
    int n = blockIdx.x * 256 + threadIdx.x;
    for (int i = threadIdx.x; i < T * 3; i += 256)
        sp[i] = path[(size_t)b * T * 3 + i];
    __syncthreads();
    if (n >= N) return;
    float sx = shape[((size_t)b * N + n) * 3 + 0];
    float sy = shape[((size_t)b * N + n) * 3 + 1];
    float sz = shape[((size_t)b * N + n) * 3 + 2];
    float acc = 0.0f;
    for (int t = 0; t < T; ++t) {
        float dx = sx - sp[t * 3 + 0];
        float dy = sy - sp[t * 3 + 1];
        float dz = sz - sp[t * 3 + 2];
        float d2 = dx * dx + dy * dy + dz * dz;
        acc += __expf(-d2 * 100.0f);
    }
    occ[(size_t)b * N + n] = fminf(fmaxf(acc, 0.0f), 1.0f);
}

// ---------------------------------------------------------------------------
// Host orchestration
// ---------------------------------------------------------------------------
extern "C" void kernel_launch(void* const* d_in, const int* in_sizes, int n_in,
                              void* d_out, int out_size, void* d_ws, size_t ws_size,
                              hipStream_t stream)
{
    (void)in_sizes; (void)n_in; (void)out_size; (void)ws_size;
    const int B = 8, T = 512, N = 2048, D = 512, H = 8, FFN = 512, L = 5;

    const float* start_pt = (const float*)d_in[0];
    const float* shape    = (const float*)d_in[1];
    const unsigned char* shape_mask = (const unsigned char*)d_in[2];
    const float* enc_w[5] = {(const float*)d_in[4], (const float*)d_in[6],
                             (const float*)d_in[8], (const float*)d_in[10],
                             (const float*)d_in[12]};
    const float* enc_b[5] = {(const float*)d_in[5], (const float*)d_in[7],
                             (const float*)d_in[9], (const float*)d_in[11],
                             (const float*)d_in[13]};
    const float* sa_wqkv = (const float*)d_in[14];
    const float* sa_bqkv = (const float*)d_in[15];
    const float* sa_wo   = (const float*)d_in[16];
    const float* sa_bo   = (const float*)d_in[17];
    const float* ca_wqkv = (const float*)d_in[18];
    const float* ca_bqkv = (const float*)d_in[19];
    const float* ca_wo   = (const float*)d_in[20];
    const float* ca_bo   = (const float*)d_in[21];
    const float* ffn_w1  = (const float*)d_in[22];
    const float* ffn_b1  = (const float*)d_in[23];
    const float* ffn_w2  = (const float*)d_in[24];
    const float* ffn_b2  = (const float*)d_in[25];
    const float* ln_g    = (const float*)d_in[26];
    const float* ln_b    = (const float*)d_in[27];
    const float* out_w   = (const float*)d_in[28];
    const float* out_b   = (const float*)d_in[29];

    // workspace layout
    float* ws = (float*)d_ws;
    size_t off = 0;
    auto alloc = [&](size_t n) { float* p = ws + off; off += n; return p; };
    float* x    = alloc((size_t)B * T * D);
    float* y    = alloc((size_t)B * T * D);
    float* att  = alloc((size_t)B * T * D);
    float* qc   = alloc((size_t)B * T * D);
    float* hb   = alloc((size_t)B * T * FFN);
    float* qkv  = alloc((size_t)B * T * 3 * D);
    float* kvc  = alloc((size_t)B * N * 2 * D);
    float* mem  = alloc((size_t)B * N * D);
    float* tA   = alloc((size_t)B * N * 128);
    float* tB   = alloc((size_t)B * N * 128);
    float* path = alloc((size_t)B * T * 3);

    const size_t gemm_shm = 2 * 64 * SAP * sizeof(_Float16);
    auto gemm = [&](const float* A, int lda, const float* W, int ldw,
                    const float* bias, const float* R, int ldr,
                    float* C, int ldc, int M, int Nn, int K, int relu) {
        dim3 g(Nn / 64, M / 64), blk(128);
        k_gemm<<<g, blk, gemm_shm, stream>>>(A, lda, W, ldw, bias, R, ldr,
                                             C, ldc, K, relu);
    };
    auto attn_shm = [&](int Lk) -> size_t {
        return (size_t)16 * Lk * 4 + (size_t)16 * Lk * 2 +
               (16 * QP + 4 * 16 * QP + 32 * QP) * sizeof(_Float16) +
               (16 * 8 + 16 + 16) * sizeof(float);
    };

    // positional encoding -> x
    k_posenc<<<dim3((T * D + 255) / 256), 256, 0, stream>>>(x, T, D, B);

    // PointNet encoder -> mem (layer 1 is a dedicated K=3 kernel)
    k_enc1<<<dim3((B * N * 64 + 255) / 256), 256, 0, stream>>>(
        shape, enc_w[0], enc_b[0], tA, B * N);
    gemm(tA, 64,  enc_w[1], 64,  enc_b[1], nullptr, 0, tB, 64,  B * N, 64,  64,  1);
    gemm(tB, 64,  enc_w[2], 64,  enc_b[2], nullptr, 0, tA, 64,  B * N, 64,  64,  1);
    gemm(tA, 64,  enc_w[3], 64,  enc_b[3], nullptr, 0, tB, 128, B * N, 128, 64,  1);
    gemm(tB, 128, enc_w[4], 128, enc_b[4], nullptr, 0, mem, D,  B * N, D,   128, 1);

    const size_t shm_self  = attn_shm(T);
    const size_t shm_cross = attn_shm(N);

    for (int i = 0; i < L; ++i) {
        // ---- self attention ----
        gemm(x, D, sa_wqkv + (size_t)i * 3 * D * D, D,
             sa_bqkv + (size_t)i * 3 * D, nullptr, 0, qkv, 3 * D, B * T, 3 * D, D, 0);
        k_attn<<<dim3(T / 16, H, B), 128, shm_self, stream>>>(
            qkv, 3 * D, qkv, 3 * D, D, 2 * D, nullptr, 1, att, D, T, T);
        gemm(att, D, sa_wo + (size_t)i * D * D, D, sa_bo + (size_t)i * D,
             x, D, y, D, B * T, D, D, 0);
        k_ln<<<B * T, 128, 0, stream>>>(y, ln_g + ((size_t)i * 3 + 0) * D,
                                        ln_b + ((size_t)i * 3 + 0) * D, x, D);
        // ---- cross attention ----
        const float* Wc = ca_wqkv + (size_t)i * 3 * D * D;
        gemm(x, D, Wc, D, ca_bqkv + (size_t)i * 3 * D, nullptr, 0,
             qc, D, B * T, D, D, 0);
        gemm(mem, D, Wc + (size_t)D * D, D, ca_bqkv + (size_t)i * 3 * D + D,
             nullptr, 0, kvc, 2 * D, B * N, 2 * D, D, 0);
        k_attn<<<dim3(T / 16, H, B), 128, shm_cross, stream>>>(
            qc, D, kvc, 2 * D, 0, D, shape_mask, 0, att, D, T, N);
        gemm(att, D, ca_wo + (size_t)i * D * D, D, ca_bo + (size_t)i * D,
             x, D, y, D, B * T, D, D, 0);
        k_ln<<<B * T, 128, 0, stream>>>(y, ln_g + ((size_t)i * 3 + 1) * D,
                                        ln_b + ((size_t)i * 3 + 1) * D, x, D);
        // ---- FFN ----
        gemm(x, D, ffn_w1 + (size_t)i * FFN * D, D, ffn_b1 + (size_t)i * FFN,
             nullptr, 0, hb, FFN, B * T, FFN, D, 1);
        gemm(hb, FFN, ffn_w2 + (size_t)i * D * FFN, FFN, ffn_b2 + (size_t)i * D,
             x, D, y, D, B * T, D, FFN, 0);
        k_ln<<<B * T, 128, 0, stream>>>(y, ln_g + ((size_t)i * 3 + 2) * D,
                                        ln_b + ((size_t)i * 3 + 2) * D, x, D);
    }

    // output head -> velocities (first part of d_out), wave-per-row GEMV
    float* vel = (float*)d_out;
    k_head<<<dim3(B * T / 4), 128, 0, stream>>>(x, out_w, out_b, vel, D);

    // path + occupancy (second part of d_out)
    k_path<<<B, 32, 0, stream>>>(vel, start_pt, path, T);
    k_occ<<<dim3(N / 256, B), 256, 0, stream>>>(
        shape, path, (float*)d_out + (size_t)B * T * 3, T, N);
}
